// AdditiveMask_76012331205217
// MI455X (gfx1250) — matrix operand order repaired and checked by reading, not compiled, not run on hardware
//
#include <hip/hip_runtime.h>
#include <math.h>

#define EPS 1e-8f

// CDNA5 has a hardware v_tanh_f32 TRANS32 op; use it if the toolchain exposes it.
#if defined(__has_builtin)
#  if __has_builtin(__builtin_amdgcn_tanhf)
#    define TANH_FAST(x) __builtin_amdgcn_tanhf(x)
#  endif
#endif
#ifndef TANH_FAST
#  define TANH_FAST(x) tanhf(x)
#endif

typedef __attribute__((ext_vector_type(16))) _Float16 v16h;
typedef __attribute__((ext_vector_type(8)))  float    v8f;

// ---------------------------------------------------------------------------
// 1) h = tanh(x @ W^T + b)   x:[N,64] f32, W:[64,64] f32, h:[N,64] f32
//    One wave32 per 16x16 output tile; K=64 as two v_wmma_f32_16x16x32_f16.
// ---------------------------------------------------------------------------
__global__ void appnp_gemm_tanh(const float* __restrict__ x,
                                const float* __restrict__ W,
                                const float* __restrict__ bias,
                                float* __restrict__ h, int N)
{
    const int wave  = threadIdx.x >> 5;           // 8 waves / block
    const int lane  = threadIdx.x & 31;
    const int tileM = blockIdx.x * 2 + (wave >> 2);
    const int tileN = wave & 3;                   // 64 cols = 4 tiles
    const int m0    = tileM * 16;
    if (m0 >= N) return;                          // wave-uniform

    const int lr = lane & 15;                     // A row / B col within tile
    const int g  = lane >> 4;                     // lane group 0/1

    const float* xr = x + (size_t)(m0 + lr) * 64;
    const float* wr = W + (size_t)(tileN * 16 + lr) * 64;  // B = W^T -> row of W

    // A 16x32 f16 layout: group g holds K in {8g..8g+7} U {16+8g..16+8g+7}
    v16h a0, a1, b0, b1;
    const int ak = g * 8;
#pragma unroll
    for (int i = 0; i < 8; ++i) {
        a0[i]     = (_Float16)xr[ak + i];
        a0[i + 8] = (_Float16)xr[ak + 16 + i];
        a1[i]     = (_Float16)xr[32 + ak + i];
        a1[i + 8] = (_Float16)xr[32 + ak + 16 + i];
    }
    // B 32x16 f16 layout: group g holds K in {8g..8g+7} U {16+8g..16+8g+7} of column lr
    const int bk = g * 8;
#pragma unroll
    for (int i = 0; i < 8; ++i) {
        b0[i]     = (_Float16)wr[bk + i];
        b0[i + 8] = (_Float16)wr[bk + 16 + i];
        b1[i]     = (_Float16)wr[32 + bk + i];
        b1[i + 8] = (_Float16)wr[32 + bk + 16 + i];
    }

    v8f c = {};
    c = __builtin_amdgcn_wmma_f32_16x16x32_f16(false, a0, false, b0,
                                               (short)0, c, false, false);
    c = __builtin_amdgcn_wmma_f32_16x16x32_f16(false, a1, false, b1,
                                               (short)0, c, false, false);

    // D layout: VGPR v, lane -> row m0 + v + 8*g, col tileN*16 + lr
    const int   col   = tileN * 16 + lr;
    const float bv    = bias[col];
    const int   mBase = m0 + g * 8;
#pragma unroll
    for (int v = 0; v < 8; ++v) {
        h[(size_t)(mBase + v) * 64 + col] = TANH_FAST(c[v] + bv);
    }
}

// ---------------------------------------------------------------------------
// 2) edge_weights[e] = relu(cos(h[src[e]], h[dst[e]])); one wave32 per edge.
// ---------------------------------------------------------------------------
__global__ void appnp_edge_cos(const float* __restrict__ h,
                               const int* __restrict__ src,
                               const int* __restrict__ dst,
                               float* __restrict__ ew, int E)
{
    const int wave = threadIdx.x >> 5;
    const int lane = threadIdx.x & 31;
    const int e = blockIdx.x * 8 + wave;
    if (e >= E) return;

    const int s = src[e];
    const int d = dst[e];
    const float2 a = ((const float2*)(h + (size_t)s * 64))[lane];
    const float2 b = ((const float2*)(h + (size_t)d * 64))[lane];

    float dot = a.x * b.x + a.y * b.y;
    float ns  = a.x * a.x + a.y * a.y;
    float nd  = b.x * b.x + b.y * b.y;
#pragma unroll
    for (int off = 16; off > 0; off >>= 1) {
        dot += __shfl_xor(dot, off, 32);
        ns  += __shfl_xor(ns,  off, 32);
        nd  += __shfl_xor(nd,  off, 32);
    }
    if (lane == 0) {
        const float n1 = fmaxf(sqrtf(ns), EPS);
        const float n2 = fmaxf(sqrtf(nd), EPS);
        const float cs = dot / (n1 * n2);
        ew[e] = cs > 0.0f ? cs : 0.0f;
    }
}

// ---------------------------------------------------------------------------
// GCN normalization helpers
// ---------------------------------------------------------------------------
__global__ void appnp_fill(float* __restrict__ p, float v, int n)
{
    const int i = blockIdx.x * blockDim.x + threadIdx.x;
    if (i < n) p[i] = v;
}

__global__ void appnp_deg_scatter(float* __restrict__ deg,
                                  const int* __restrict__ dst,
                                  const float* __restrict__ ew, int E)
{
    const int e = blockIdx.x * blockDim.x + threadIdx.x;
    if (e < E) atomicAdd(&deg[dst[e]], ew[e]);
}

__global__ void appnp_dis(float* __restrict__ deg_to_dis, int N)
{
    const int i = blockIdx.x * blockDim.x + threadIdx.x;
    if (i < N) {
        const float dg = deg_to_dis[i];
        deg_to_dis[i] = dg > 0.0f ? rsqrtf(fmaxf(dg, EPS)) : 0.0f;
    }
}

__global__ void appnp_normw(float* __restrict__ normw,
                            const float* __restrict__ dis,
                            const int* __restrict__ src,
                            const int* __restrict__ dst,
                            const float* __restrict__ ew, int E)
{
    const int e = blockIdx.x * blockDim.x + threadIdx.x;
    if (e < E) normw[e] = dis[src[e]] * ew[e] * dis[dst[e]];
}

// ---------------------------------------------------------------------------
// APPNP propagation
// ---------------------------------------------------------------------------
__global__ void appnp_f0(float* __restrict__ f0, float* __restrict__ f,
                         const float* __restrict__ mask, int N)
{
    const int i = blockIdx.x * blockDim.x + threadIdx.x;
    if (i < N) {
        const float v = fmaxf(mask[i], 0.0f);
        f0[i] = v;
        f[i]  = v;
    }
}

__global__ void appnp_agg_init(float* __restrict__ agg,
                               const float* __restrict__ dis,
                               const float* __restrict__ f, int N)
{
    const int i = blockIdx.x * blockDim.x + threadIdx.x;
    if (i < N) {
        const float d = dis[i];              // self loop: weight dis[i]^2 * f[i]
        agg[i] = d * d * f[i];
    }
}

__global__ void appnp_scatter(float* __restrict__ agg,
                              const float* __restrict__ normw,
                              const float* __restrict__ f,
                              const int* __restrict__ src,
                              const int* __restrict__ dst, int E)
{
    const int e = blockIdx.x * blockDim.x + threadIdx.x;
    if (e < E) atomicAdd(&agg[dst[e]], normw[e] * f[src[e]]);
}

__global__ void appnp_update(float* __restrict__ f,
                             const float* __restrict__ agg,
                             const float* __restrict__ f0,
                             const float* __restrict__ alpha_p, int N)
{
    const int i = blockIdx.x * blockDim.x + threadIdx.x;
    if (i < N) {
        const float a = *alpha_p;
        f[i] = (1.0f - a) * agg[i] + a * f0[i];
    }
}

// ---------------------------------------------------------------------------
extern "C" void kernel_launch(void* const* d_in, const int* in_sizes, int n_in,
                              void* d_out, int out_size, void* d_ws, size_t ws_size,
                              hipStream_t stream)
{
    const float* x     = (const float*)d_in[0];
    const float* mask  = (const float*)d_in[1];
    const int*   edge  = (const int*)d_in[2];
    const float* W     = (const float*)d_in[3];
    const float* bias  = (const float*)d_in[4];
    const float* alpha = (const float*)d_in[5];

    const int N = in_sizes[0] / 64;
    const int E = in_sizes[2] / 2;
    const int* src = edge;
    const int* dst = edge + E;

    float* out_f  = (float*)d_out;       // [N] : propagated features (f)
    float* out_ew = (float*)d_out + N;   // [E] : edge weights

    char*  ws    = (char*)d_ws;
    float* h     = (float*)ws;  ws += (size_t)N * 64 * sizeof(float);
    float* dis   = (float*)ws;  ws += (size_t)N * sizeof(float);
    float* normw = (float*)ws;  ws += (size_t)E * sizeof(float);
    float* f0    = (float*)ws;  ws += (size_t)N * sizeof(float);
    float* agg   = (float*)ws;  ws += (size_t)N * sizeof(float);

    const int T = 256;

    // 1) h = tanh(x W^T + b)  (WMMA)
    appnp_gemm_tanh<<<(N + 31) / 32, T, 0, stream>>>(x, W, bias, h, N);

    // 2) edge cosine-similarity weights (one wave per edge, L2-resident h)
    appnp_edge_cos<<<(E + 7) / 8, T, 0, stream>>>(h, src, dst, out_ew, E);

    // 3) gcn_norm: deg (self-loop folded in as init=1), dis = rsqrt, norm_w
    appnp_fill<<<(N + T - 1) / T, T, 0, stream>>>(dis, 1.0f, N);   // deg in-place
    appnp_deg_scatter<<<(E + T - 1) / T, T, 0, stream>>>(dis, dst, out_ew, E);
    appnp_dis<<<(N + T - 1) / T, T, 0, stream>>>(dis, N);
    appnp_normw<<<(E + T - 1) / T, T, 0, stream>>>(normw, dis, src, dst, out_ew, E);

    // 4) APPNP: f0 = relu(mask), K=5 propagation steps (f lives in d_out)
    appnp_f0<<<(N + T - 1) / T, T, 0, stream>>>(f0, out_f, mask, N);
    for (int k = 0; k < 5; ++k) {
        appnp_agg_init<<<(N + T - 1) / T, T, 0, stream>>>(agg, dis, out_f, N);
        appnp_scatter<<<(E + T - 1) / T, T, 0, stream>>>(agg, normw, out_f, src, dst, E);
        appnp_update<<<(N + T - 1) / T, T, 0, stream>>>(out_f, agg, f0, alpha, N);
    }
}
